// BitNetAttention_6803228196889
// MI455X (gfx1250) — matrix operand order repaired and checked by reading, not compiled, not run on hardware
//
#include <hip/hip_runtime.h>
#include <hip/hip_bf16.h>
#include <stdint.h>

// ---------------------------------------------------------------------------
// BitNet GQA attention for MI455X (gfx1250).
// - All matmuls: v_wmma_f32_16x16x32_bf16 (wave32, f32 accumulate)
// - GEMM operand slabs: Tensor Data Mover (tensor_load_to_lds, TENSORcnt)
// - Attention K/V tiles: global_load_async_to_lds_b128 (ASYNCcnt)
// Both staged double-buffered in LDS.
// ---------------------------------------------------------------------------

#define DEVI __device__ __forceinline__

typedef __attribute__((ext_vector_type(16))) __bf16 v16bf;
typedef __attribute__((ext_vector_type(8)))  __bf16 v8bf;
typedef __attribute__((ext_vector_type(8)))  float  v8f;
typedef __attribute__((ext_vector_type(4)))  unsigned int u32x4;
typedef __attribute__((ext_vector_type(8)))  int i32x8;
typedef __attribute__((ext_vector_type(4)))  int i32x4;

constexpr int B_   = 2;
constexpr int S_   = 2048;
constexpr int HID_ = 2560;
constexpr int NH_  = 20;
constexpr int NKV_ = 5;
constexpr int HD_  = 128;
constexpr int G_   = NH_ / NKV_;       // 4
constexpr float THETA_ = 500000.0f;
constexpr int M_ROWS = B_ * S_;        // 4096 flat (b,s) rows

// ---------------- CDNA5 async copy: global -> LDS (ASYNCcnt) ----------------

// One lane moves 16 bytes: LDS[lds] = MEM[gptr .. gptr+15]  (GV mode)
DEVI void async_load_b128(const __bf16* gptr, const __bf16* lptr) {
  const uint32_t lds_addr = (uint32_t)(uintptr_t)lptr;   // low 32b = LDS offset
  const uint64_t gaddr    = (uint64_t)(uintptr_t)gptr;
  asm volatile("global_load_async_to_lds_b128 %0, %1, off"
               :: "v"(lds_addr), "v"(gaddr)
               : "memory");
}

template <int N>
DEVI void s_wait_async() {
  asm volatile("s_wait_asynccnt %0" :: "i"(N) : "memory");
}

// ------------- CDNA5 Tensor Data Mover: 2D tile global -> LDS ---------------
// D# per CDNA5 ISA: group0 = {flags, lds_addr, gaddr_lo,
// gaddr_hi|type}, group1 = dims/strides. data_size = 2 bytes (bf16).
// Issue from ONE wave only (EXEC ignored, SGPR descriptor); track TENSORcnt.
DEVI void tdm_load_2d(const __bf16* gptr, const __bf16* lptr,
                      int rows, int cols, int stride_elems) {
  const uint64_t ga = (uint64_t)(uintptr_t)gptr;
  u32x4 g0;
  g0[0] = 1u;                                        // count=1, user mode
  g0[1] = (uint32_t)(uintptr_t)lptr;                 // lds_addr [63:32]
  g0[2] = (uint32_t)(ga & 0xffffffffu);              // global_addr lo
  g0[3] = (uint32_t)((ga >> 32) & 0x01ffffffu)       // global_addr [56:32]
        | (2u << 30);                                // type = 2 ("image")
  const int td0 = stride_elems;                      // tensor_dim0 (>= cols)
  const int td1 = rows;                              // tensor_dim1
  i32x8 g1;
  g1[0] = 0x00010000;                                // data_size=1 (2 bytes)
  g1[1] = (td0 & 0xffff) << 16;                      // tensor_dim0 [63:48]
  g1[2] = ((td0 >> 16) & 0xffff) | ((td1 & 0xffff) << 16);   // [79:64],[95:80]
  g1[3] = ((td1 >> 16) & 0xffff) | ((cols & 0xffff) << 16);  // tile_dim0
  g1[4] = rows & 0xffff;                             // tile_dim1 (tile_dim2=0)
  g1[5] = stride_elems;                              // tensor_dim0_stride lo32
  g1[6] = 0;                                         // stride hi / dim1_stride
  g1[7] = 0;
  i32x4 gz = {0, 0, 0, 0};
#if defined(__clang_major__) && (__clang_major__ >= 23)
  i32x8 gz8 = {0, 0, 0, 0, 0, 0, 0, 0};
  __builtin_amdgcn_tensor_load_to_lds(g0, g1, gz, gz, gz8, 0);
#else
  __builtin_amdgcn_tensor_load_to_lds(g0, g1, gz, gz, 0);
#endif
}

template <int N>
DEVI void s_wait_tensor() {
  __builtin_amdgcn_s_wait_tensorcnt((short)N);
}

// ---------------- WMMA fragment helpers (ISA 7.12.2 layouts) ----------------

// A-matrix 16x32 bf16, row-major source, leading dim `lda` (elements).
DEVI v16bf load_a_frag(const __bf16* A, int lda) {
  const int lane = threadIdx.x & 31;
  const int hf   = lane >> 4;
  const __bf16* row = A + (size_t)(lane & 15) * lda + 8 * hf;
  v16bf f;
  ((v8bf*)&f)[0] = *(const v8bf*)(row);
  ((v8bf*)&f)[1] = *(const v8bf*)(row + 16);
  return f;
}

// B-matrix (K=32,N=16): B[k][n] = Bm[n][k], Bm row-major, ld `ldb`.
DEVI v16bf load_b_frag(const __bf16* Bm, int ldb) {
  const int lane = threadIdx.x & 31;
  const __bf16* row = Bm + (size_t)(lane & 15) * ldb + 8 * (lane >> 4);
  v16bf f;
  ((v8bf*)&f)[0] = *(const v8bf*)(row);
  ((v8bf*)&f)[1] = *(const v8bf*)(row + 16);
  return f;
}

DEVI v8f wmma_bf16(v16bf a, v16bf b, v8f c) {
  return __builtin_amdgcn_wmma_f32_16x16x32_bf16(
      false, a, false, b, (short)0, c, false, false);
}

// ---------------------------- fp32 -> bf16 copy -----------------------------

__global__ void cvt_f32_bf16(const float* __restrict__ in,
                             __bf16* __restrict__ out, unsigned n) {
  unsigned i = blockIdx.x * 256u + threadIdx.x;
  if (i < n) out[i] = (__bf16)in[i];
}

// ------------------------------- WMMA GEMM ----------------------------------
// C[M,N] = A[M,K] * W[N,K]^T.  256 thr = 8 waves.  Block tile 128M x 128N,
// k-step 64.  Wave tile 32M x 64N.  A/B k-slabs double-buffered in LDS via
// TDM descriptors (one per slab, issued by wave 0, TENSORcnt-tracked).
// MODE 0: bf16 row-major   MODE 1: bf16 V-transposed [B,NKV,HD,S]   MODE 2: f32
template <int MODE>
__global__ __launch_bounds__(256) void gemm_wmma(
    const __bf16* __restrict__ A, const __bf16* __restrict__ W,
    void* __restrict__ Cout, int N, int K) {
  __shared__ __bf16 At[2][128 * 64];   // [m][k] slab, 16 KB each
  __shared__ __bf16 Bt[2][128 * 64];   // [n][k] slab

  const int tid  = threadIdx.x;
  const int lane = tid & 31;
  const int wave = tid >> 5;
  const int hf   = lane >> 4;
  const int nn   = lane & 15;
  const int wm   = wave & 3;           // M subtile 0..3
  const int wn   = wave >> 2;          // N subtile 0..1
  const int m0   = blockIdx.y * 128;
  const int n0   = blockIdx.x * 128;

  auto issue = [&](int buf, int kk) {
    if (wave == 0) {
      tdm_load_2d(A + (size_t)m0 * K + kk, &At[buf][0], 128, 64, K);
      tdm_load_2d(W + (size_t)n0 * K + kk, &Bt[buf][0], 128, 64, K);
    }
  };

  const v8f zro = {0.f, 0.f, 0.f, 0.f, 0.f, 0.f, 0.f, 0.f};
  v8f acc[2][4] = {{zro, zro, zro, zro}, {zro, zro, zro, zro}};

  const int nsteps = K >> 6;
  issue(0, 0);
  for (int step = 0; step < nsteps; ++step) {
    const int cur = step & 1;
    if (step + 1 < nsteps) {
      issue(cur ^ 1, (step + 1) << 6);
      if (wave == 0) s_wait_tensor<2>();   // current slab landed
    } else {
      if (wave == 0) s_wait_tensor<0>();
    }
    __syncthreads();

#pragma unroll
    for (int kc = 0; kc < 64; kc += 32) {
      // batch fragment loads so one DS wait covers the group
      v16bf a0 = load_a_frag(&At[cur][(wm * 32 +  0) * 64 + kc], 64);
      v16bf a1 = load_a_frag(&At[cur][(wm * 32 + 16) * 64 + kc], 64);
      v16bf bf[4];
#pragma unroll
      for (int j = 0; j < 4; ++j)
        bf[j] = load_b_frag(&Bt[cur][(wn * 64 + 16 * j) * 64 + kc], 64);
#pragma unroll
      for (int j = 0; j < 4; ++j) {
        acc[0][j] = wmma_bf16(a0, bf[j], acc[0][j]);
        acc[1][j] = wmma_bf16(a1, bf[j], acc[1][j]);
      }
    }
    __syncthreads();
  }

#pragma unroll
  for (int mi = 0; mi < 2; ++mi) {
#pragma unroll
    for (int j = 0; j < 4; ++j) {
#pragma unroll
      for (int r = 0; r < 8; ++r) {
        const int m = m0 + wm * 32 + mi * 16 + r + 8 * hf;
        const int n = n0 + wn * 64 + 16 * j + nn;
        const float v = acc[mi][j][r];
        if (MODE == 0) {
          ((__bf16*)Cout)[(size_t)m * N + n] = (__bf16)v;
        } else if (MODE == 1) {
          const int b  = m >> 11;            // / S_
          const int s  = m & (S_ - 1);
          const int kv = n >> 7;             // / HD_
          const int hd = n & (HD_ - 1);
          ((__bf16*)Cout)[(((size_t)(b * NKV_ + kv)) * HD_ + hd) * S_ + s] =
              (__bf16)v;
        } else {
          ((float*)Cout)[(size_t)m * N + n] = v;
        }
      }
    }
  }
}

// ------------------------- RoPE + head rearrange ----------------------------
__global__ void rope_rearrange(const __bf16* __restrict__ in,
                               __bf16* __restrict__ out, int H) {
  size_t idx = (size_t)blockIdx.x * 256 + threadIdx.x;  // over B*S*H*64
  const int d = (int)(idx & 63);
  size_t t = idx >> 6;
  const int h = (int)(t % H); t /= H;
  const int s = (int)(t % S_);
  const int b = (int)(t / S_);

  const __bf16* p = in + (((size_t)(b * S_ + s)) * H + h) * HD_;
  const float x1 = (float)p[d];
  const float x2 = (float)p[d + 64];
  const float inv = __expf(-((float)(2 * d) / (float)HD_) * __logf(THETA_));
  const float ang = (float)s * inv;
  float c, sn;
  __sincosf(ang, &c, &sn);

  __bf16* q = out + (((size_t)(b * H + h)) * S_ + s) * HD_;
  q[d]      = (__bf16)(x1 * c - x2 * sn);
  q[d + 64] = (__bf16)(x2 * c + x1 * sn);
}

// ----------------------- flash-style causal attention -----------------------
// grid: (S/64, B*NH), block 128 (4 waves). Wave owns 16 query rows.
// K/V tiles double-buffered in LDS via async copies (ASYNCcnt).
__global__ __launch_bounds__(128) void attn_wmma(
    const __bf16* __restrict__ Q, const __bf16* __restrict__ Kc,
    const __bf16* __restrict__ Vt, const float* __restrict__ sub_w,
    __bf16* __restrict__ out) {
  __shared__ __bf16 Ktile[2][64 * 128];  // [t][hd], 16 KB each
  __shared__ __bf16 Vtile[2][128 * 64];  // [hd][t], 16 KB each
  __shared__ __bf16 pbuf[4][16 * 64];    // per-wave P strip

  const int tid  = threadIdx.x;
  const int lane = tid & 31;
  const int wave = tid >> 5;
  const int hf   = lane >> 4;
  const int nn   = lane & 15;

  const int bh  = blockIdx.y;            // = b*NH + h
  const int b   = bh / NH_;
  const int h   = bh % NH_;
  const int kvh = h / G_;
  const int q0  = blockIdx.x * 64 + wave * 16;

  const __bf16* Qh = Q  + (size_t)bh * S_ * HD_;
  const __bf16* Kh = Kc + (size_t)(b * NKV_ + kvh) * S_ * HD_;
  const __bf16* Vh = Vt + (size_t)(b * NKV_ + kvh) * HD_ * S_;

  auto issueKV = [&](int buf, int t0) {
#pragma unroll
    for (int i = 0; i < 8; ++i) {
      const int idx = tid + i * 128;       // 0..1023, 8 elements each
      const int kr = idx >> 4, kcol = (idx & 15) * 8;   // K: [64][128]
      async_load_b128(Kh + (size_t)(t0 + kr) * HD_ + kcol,
                      &Ktile[buf][kr * 128 + kcol]);
      const int vr = idx >> 3, vcol = (idx & 7) * 8;    // V: [128][64]
      async_load_b128(Vh + (size_t)vr * S_ + t0 + vcol,
                      &Vtile[buf][vr * 64 + vcol]);
    }
  };

  // Q fragments for this wave's 16 rows (HD=128 -> 4 k-chunks)
  v16bf aq[4];
#pragma unroll
  for (int kc = 0; kc < 4; ++kc)
    aq[kc] = load_a_frag(Qh + (size_t)q0 * HD_ + 32 * kc, HD_);

  const v8f zro = {0.f, 0.f, 0.f, 0.f, 0.f, 0.f, 0.f, 0.f};
  v8f oacc[8] = {zro, zro, zro, zro, zro, zro, zro, zro};
  float rowm[8], rowl[8];
#pragma unroll
  for (int r = 0; r < 8; ++r) { rowm[r] = -3.0e38f; rowl[r] = 0.f; }

  const float sc = 0.08838834764831845f;   // 1/sqrt(128)
  const int ktiles = blockIdx.x + 1;       // uniform across block (causal)

  issueKV(0, 0);
  for (int kt = 0; kt < ktiles; ++kt) {
    const int cur = kt & 1;
    const int t0  = kt * 64;
    if (kt + 1 < ktiles) {
      issueKV(cur ^ 1, (kt + 1) * 64);
      s_wait_async<16>();
    } else {
      s_wait_async<0>();
    }
    __syncthreads();

    // ---- scores S = Q * K^T for 16q x 64t tile (from LDS) ----
    v8f sacc[4] = {zro, zro, zro, zro};
#pragma unroll
    for (int nt = 0; nt < 4; ++nt) {
      v16bf bk[4];
#pragma unroll
      for (int kc = 0; kc < 4; ++kc)
        bk[kc] = load_b_frag(&Ktile[cur][(16 * nt) * 128 + 32 * kc], 128);
#pragma unroll
      for (int kc = 0; kc < 4; ++kc)
        sacc[nt] = wmma_bf16(aq[kc], bk[kc], sacc[nt]);
    }

    // ---- scale + causal mask (C layout: row=r+8*hf, col=nn) ----
#pragma unroll
    for (int nt = 0; nt < 4; ++nt) {
#pragma unroll
      for (int r = 0; r < 8; ++r) {
        const int tcol = t0 + 16 * nt + nn;
        const int qrow = q0 + r + 8 * hf;
        const float v = sacc[nt][r] * sc;
        sacc[nt][r] = (tcol <= qrow) ? v : -3.0e38f;
      }
    }

    // ---- online softmax per row (reduce across 16 lanes of the half) ----
#pragma unroll
    for (int r = 0; r < 8; ++r) {
      float mloc = -3.0e38f;
#pragma unroll
      for (int nt = 0; nt < 4; ++nt) mloc = fmaxf(mloc, sacc[nt][r]);
#pragma unroll
      for (int off = 8; off >= 1; off >>= 1)
        mloc = fmaxf(mloc, __shfl_xor(mloc, off, 32));
      const float nm    = fmaxf(rowm[r], mloc);
      const float alpha = __expf(rowm[r] - nm);
      float rs = 0.f;
#pragma unroll
      for (int nt = 0; nt < 4; ++nt) {
        const float p = __expf(sacc[nt][r] - nm);
        sacc[nt][r] = p;
        rs += p;
      }
#pragma unroll
      for (int off = 8; off >= 1; off >>= 1) rs += __shfl_xor(rs, off, 32);
      rowl[r] = rowl[r] * alpha + rs;
      rowm[r] = nm;
#pragma unroll
      for (int ot = 0; ot < 8; ++ot) oacc[ot][r] *= alpha;
    }

    // ---- C-layout -> A-layout reshape through per-wave LDS strip ----
#pragma unroll
    for (int nt = 0; nt < 4; ++nt)
#pragma unroll
      for (int r = 0; r < 8; ++r)
        pbuf[wave][(r + 8 * hf) * 64 + 16 * nt + nn] = (__bf16)sacc[nt][r];

    // ---- O += P * V  (2 key-chunks of 32, 8 hd-tiles of 16, from LDS) ----
#pragma unroll
    for (int kc2 = 0; kc2 < 2; ++kc2) {
      v16bf ap = load_a_frag(&pbuf[wave][32 * kc2], 64);
#pragma unroll
      for (int og = 0; og < 2; ++og) {          // groups of 4 hd-tiles
        v16bf bv[4];
#pragma unroll
        for (int j = 0; j < 4; ++j)
          bv[j] = load_b_frag(
              &Vtile[cur][(16 * (og * 4 + j)) * 64 + 32 * kc2], 64);
#pragma unroll
        for (int j = 0; j < 4; ++j)
          oacc[og * 4 + j] = wmma_bf16(ap, bv[j], oacc[og * 4 + j]);
      }
    }
    __syncthreads();
  }

  // ---- normalize, apply BitNetSubNorm gain, store bf16 [B,S,NH*HD] ----
#pragma unroll
  for (int ot = 0; ot < 8; ++ot) {
#pragma unroll
    for (int r = 0; r < 8; ++r) {
      const int qrow = q0 + r + 8 * hf;
      const int ch   = h * HD_ + 16 * ot + nn;
      const float v  = oacc[ot][r] / rowl[r];
      out[((size_t)(b * S_ + qrow)) * (NH_ * HD_) + ch] =
          (__bf16)(v * sub_w[ch]);
    }
  }
}

// --------------------------------- launch -----------------------------------

extern "C" void kernel_launch(void* const* d_in, const int* in_sizes, int n_in,
                              void* d_out, int out_size, void* d_ws,
                              size_t ws_size, hipStream_t stream) {
  const float* x_f32  = (const float*)d_in[0];  // [B,S,HID]
  // d_in[1] = position_ids (== arange(S) per batch) -> implicit, unused
  const float* wq_f32 = (const float*)d_in[2];  // [NH*HD, HID]
  const float* wk_f32 = (const float*)d_in[3];  // [NKV*HD, HID]
  const float* wv_f32 = (const float*)d_in[4];  // [NKV*HD, HID]
  const float* wo_f32 = (const float*)d_in[5];  // [HID, NH*HD]
  const float* sub_w  = (const float*)d_in[6];  // [NH*HD]

  constexpr size_t N_X  = (size_t)B_ * S_ * HID_;
  constexpr size_t N_WQ = (size_t)NH_ * HD_ * HID_;
  constexpr size_t N_WK = (size_t)NKV_ * HD_ * HID_;
  constexpr size_t N_WV = N_WK;
  constexpr size_t N_WO = (size_t)HID_ * NH_ * HD_;
  constexpr size_t N_Q  = (size_t)B_ * S_ * NH_ * HD_;
  constexpr size_t N_K  = (size_t)B_ * S_ * NKV_ * HD_;
  constexpr size_t N_V  = N_K;

  __bf16* p = (__bf16*)d_ws;
  __bf16* Xb  = p; p += N_X;
  __bf16* Wqb = p; p += N_WQ;
  __bf16* Wkb = p; p += N_WK;
  __bf16* Wvb = p; p += N_WV;
  __bf16* Wob = p; p += N_WO;
  __bf16* Qa  = p; p += N_Q;   // pre-RoPE Q [B,S,NH,HD]; reused as attn out
  __bf16* Ka  = p; p += N_K;   // pre-RoPE K [B,S,NKV,HD]
  __bf16* Vt  = p; p += N_V;   // V transposed [B,NKV,HD,S]
  __bf16* Qr  = p; p += N_Q;   // [B,NH,S,HD]
  __bf16* Kr  = p; p += N_K;   // [B,NKV,S,HD]
  __bf16* Attn = Qa;           // [B,S,NH*HD] (Qa dead after RoPE)

  auto cvt = [&](const float* src, __bf16* dst, size_t n) {
    cvt_f32_bf16<<<dim3((unsigned)((n + 255) / 256)), dim3(256), 0, stream>>>(
        src, dst, (unsigned)n);
  };
  cvt(x_f32, Xb, N_X);
  cvt(wq_f32, Wqb, N_WQ);
  cvt(wk_f32, Wkb, N_WK);
  cvt(wv_f32, Wvb, N_WV);
  cvt(wo_f32, Wob, N_WO);

  // QKV projections: C[M=4096, N] = X * W^T, K=2560
  gemm_wmma<0><<<dim3(NH_ * HD_ / 128, M_ROWS / 128), dim3(256), 0, stream>>>(
      Xb, Wqb, (void*)Qa, NH_ * HD_, HID_);
  gemm_wmma<0><<<dim3(NKV_ * HD_ / 128, M_ROWS / 128), dim3(256), 0, stream>>>(
      Xb, Wkb, (void*)Ka, NKV_ * HD_, HID_);
  gemm_wmma<1><<<dim3(NKV_ * HD_ / 128, M_ROWS / 128), dim3(256), 0, stream>>>(
      Xb, Wvb, (void*)Vt, NKV_ * HD_, HID_);

  // RoPE + [B,S,H,HD] -> [B,H,S,HD]
  rope_rearrange<<<dim3((unsigned)((size_t)B_ * S_ * NH_ * 64 / 256)),
                   dim3(256), 0, stream>>>(Qa, Qr, NH_);
  rope_rearrange<<<dim3((unsigned)((size_t)B_ * S_ * NKV_ * 64 / 256)),
                   dim3(256), 0, stream>>>(Ka, Kr, NKV_);

  // causal GQA attention (+ sub_w gain), bf16 out [B,S,NH*HD]
  attn_wmma<<<dim3(S_ / 64, B_ * NH_), dim3(128), 0, stream>>>(
      Qr, Kr, Vt, sub_w, Attn);

  // o_proj: fp32 out = Attn * Wo^T
  gemm_wmma<2><<<dim3(HID_ / 128, M_ROWS / 128), dim3(256), 0, stream>>>(
      Attn, Wob, d_out, HID_, NH_ * HD_);
}
